// MaskedCrossAttention_27986006901343
// MI455X (gfx1250) — matrix operand order repaired and checked by reading, not compiled, not run on hardware
//
#include <hip/hip_runtime.h>
#include <hip/hip_bf16.h>
#include <stdint.h>

// ---------------------------------------------------------------------------
// MaskedCrossAttention for MI455X (gfx1250, wave32, WMMA bf16 + TDM staging)
// ---------------------------------------------------------------------------

typedef __attribute__((ext_vector_type(16))) __bf16 v16bf;
typedef __attribute__((ext_vector_type(8)))  float  v8f;

constexpr int Bc   = 4;
constexpr int TQ   = 2048;
constexpr int DIM  = 1024;
constexpr int NLAT = 64;
constexpr int DLAT = 1024;
constexpr int H    = 8;
constexpr int D    = 64;
constexpr int Jtot = 1024;         // TKV * NLAT
constexpr int HD   = H * D;        // 512

#if defined(__has_builtin)
#  if __has_builtin(__builtin_amdgcn_tensor_load_to_lds)
#    define USE_TDM 1
#  endif
#endif
#ifndef USE_TDM
#  define USE_TDM 0
#endif

union Frag {
  v16bf  v;
  uint4  q[2];
  __bf16 b[16];
};

__device__ __forceinline__ __bf16 f2bf(float f) {
  union { float f; uint32_t u; } c; c.f = f;
  uint32_t u = c.u;
  uint32_t r = (u + 0x7FFFu + ((u >> 16) & 1u)) >> 16;   // round-to-nearest-even
  union { uint16_t s; __bf16 b; } o; o.s = (uint16_t)r;
  return o.b;
}

__device__ __forceinline__ v8f zero8() {
  v8f z;
#pragma unroll
  for (int i = 0; i < 8; ++i) z[i] = 0.0f;
  return z;
}

__device__ __forceinline__ v16bf frag_ld(const __bf16* p0, const __bf16* p1) {
  Frag f;
  f.q[0] = *(const uint4*)p0;
  f.q[1] = *(const uint4*)p1;
  return f.v;
}

// A-matrix fragment (16x32 bf16) from row-major [M][K] memory (ISA 7.12.2 layout).
__device__ __forceinline__ v16bf a_frag(const __bf16* A, int lda, int m0, int k0, int lane) {
  const __bf16* r = A + (size_t)(m0 + (lane & 15)) * lda + k0 + ((lane >> 4) << 3);
  return frag_ld(r, r + 16);
}

// B-matrix fragment (32x16 bf16) from [N][K] row-major (i.e. B^T) memory.
__device__ __forceinline__ v16bf b_frag(const __bf16* Bt, int ldb, int n0, int k0, int lane) {
  const __bf16* r = Bt + (size_t)(n0 + (lane & 15)) * ldb + k0 + ((lane >> 4) << 3);
  return frag_ld(r, r + 16);
}

__device__ __forceinline__ v8f wmma_bf16(v16bf a, v16bf b, v8f c) {
  return __builtin_amdgcn_wmma_f32_16x16x32_bf16(false, a, false, b, (short)0, c, false, false);
}

// ---------------------------------------------------------------------------
// TDM: stage a 64-row x 64-element bf16 tile (row stride = row_stride elems)
// from global into LDS.  2-D descriptor per CDNA5 ISA.
// ---------------------------------------------------------------------------
#if USE_TDM
typedef __attribute__((ext_vector_type(4))) unsigned int u32x4;
typedef __attribute__((ext_vector_type(8))) int i32x8;
typedef __attribute__((ext_vector_type(4))) int i32x4;

__device__ __forceinline__ void tdm_issue(uint32_t lds_addr, const __bf16* gsrc,
                                          int tensor_rows, int tensor_cols, int row_stride) {
  uint64_t ga = (uint64_t)(size_t)gsrc;
  // Group 0: count=1 | lds_addr | global_addr[56:0] | type=2
  u32x4 g0 = { 1u, lds_addr, (uint32_t)ga,
               (uint32_t)((ga >> 32) & 0x01FFFFFFu) | 0x80000000u };
  // Group 1: data_size=2B; tensor_dim0/1; tile 64x64; dim0 stride
  uint32_t tc = (uint32_t)tensor_cols, tr = (uint32_t)tensor_rows;
  i32x8 g1 = {
    (int)(1u << 16),                                   // [17:16] data_size = 1 (2 bytes)
    (int)((tc & 0xFFFFu) << 16),                       // tensor_dim0 lo16
    (int)(((tc >> 16) & 0xFFFFu) | ((tr & 0xFFFFu) << 16)),
    (int)(((tr >> 16) & 0xFFFFu) | (64u << 16)),       // tile_dim0 = 64
    (int)64,                                           // tile_dim1 = 64, tile_dim2 = 0
    (int)row_stride,                                   // tensor_dim0_stride lo32
    0, 0
  };
  i32x4 gz = {0, 0, 0, 0};
#if defined(__clang_major__) && __clang_major__ >= 23
  i32x8 gz8 = {0, 0, 0, 0, 0, 0, 0, 0};
  __builtin_amdgcn_tensor_load_to_lds(g0, g1, gz, gz, gz8, 0);
#else
  __builtin_amdgcn_tensor_load_to_lds(g0, g1, gz, gz, 0);
#endif
}
#endif

// ---------------------------------------------------------------------------
// Shared GEMM inner pipeline.
// Block = 8 waves on one 64-column weight panel; wave w covers rows
// m0 = blockRow*128 + w*16.  Weight panel double-buffered in LDS with
// STATIC buffer indices (chunk loop unrolled by 2 -> ds_load, not flat),
// TDM stages chunk i+1 while chunk i feeds the WMMAs; A frags ping-pong
// between even/odd register sets (no copies).  K/64 must be even (16/16/8).
// ---------------------------------------------------------------------------
__device__ __forceinline__ void gemm_panel(const __bf16* __restrict__ A, int lda, int m0,
                                           const __bf16* __restrict__ Wt, int K, int N, int n0,
                                           int w, int lane, v8f acc[4]) {
  __shared__ __bf16 Bsh[2][64 * 64];
  const int nch = K >> 6;
  const __bf16* Wp = Wt + (size_t)n0 * K;

#if !USE_TDM
  auto stage = [&](int buf, int i) {    // cooperative fallback copy (64 x 128B)
    for (int t = threadIdx.x; t < 512; t += 256) {
      int r = t >> 3, c = t & 7;
      *(uint4*)(&Bsh[buf][r * 64 + c * 8]) =
          *(const uint4*)(Wp + (size_t)r * K + i * 64 + c * 8);
    }
  };
#endif

  // compute one 64-deep chunk: hoist all 8 B frags (ds_load clause), then 8 WMMAs
  auto compute = [&](int buf, v16bf aa0, v16bf aa1) {
    v16bf bb[8];
#pragma unroll
    for (int f = 0; f < 4; ++f) {
      bb[2 * f]     = b_frag(&Bsh[buf][0], 64, f * 16, 0, lane);
      bb[2 * f + 1] = b_frag(&Bsh[buf][0], 64, f * 16, 32, lane);
    }
#pragma unroll
    for (int f = 0; f < 4; ++f) {
      acc[f] = wmma_bf16(aa0, bb[2 * f], acc[f]);
      acc[f] = wmma_bf16(aa1, bb[2 * f + 1], acc[f]);
    }
  };

#if USE_TDM
  if (w == 0) tdm_issue((uint32_t)(size_t)&Bsh[0][0], Wp, N, K, K);
#else
  stage(0, 0);
#endif
  v16bf a0e = a_frag(A, lda, m0, 0, lane);
  v16bf a1e = a_frag(A, lda, m0, 32, lane);
  v16bf a0o = a0e, a1o = a1e;

  for (int i = 0; i < nch; i += 2) {
    // ---- even chunk i (buffer 0) ----
#if USE_TDM
    if (w == 0) {
      tdm_issue((uint32_t)(size_t)&Bsh[1][0], Wp + (i + 1) * 64, N, K, K);
      __builtin_amdgcn_s_wait_tensorcnt(1);    // chunk i landed, i+1 in flight
    }
    __syncthreads();
#else
    __syncthreads();
#endif
    a0o = a_frag(A, lda, m0, (i + 1) * 64, lane);        // prefetch odd A
    a1o = a_frag(A, lda, m0, (i + 1) * 64 + 32, lane);
    compute(0, a0e, a1e);
    __syncthreads();                                     // buf0 reads done

    // ---- odd chunk i+1 (buffer 1) ----
#if USE_TDM
    if (w == 0) {
      if (i + 2 < nch) {
        tdm_issue((uint32_t)(size_t)&Bsh[0][0], Wp + (i + 2) * 64, N, K, K);
        __builtin_amdgcn_s_wait_tensorcnt(1);  // chunk i+1 landed, i+2 in flight
      } else {
        __builtin_amdgcn_s_wait_tensorcnt(0);
      }
    }
    __syncthreads();
#else
    stage(1, i + 1);
    __syncthreads();
#endif
    if (i + 2 < nch) {
      a0e = a_frag(A, lda, m0, (i + 2) * 64, lane);      // prefetch even A
      a1e = a_frag(A, lda, m0, (i + 2) * 64 + 32, lane);
    }
    compute(1, a0o, a1o);
    __syncthreads();                                     // buf1 reads done
#if !USE_TDM
    if (i + 2 < nch) stage(0, i + 2);
#endif
  }
}

// ---------------------------------------------------------------------------
// Prep kernels
// ---------------------------------------------------------------------------

__global__ __launch_bounds__(256) void k_qtime(const uint8_t* __restrict__ media,
                                               int* __restrict__ qt) {
  __shared__ int s[TQ];
  int b = blockIdx.x;
  for (int i = threadIdx.x; i < TQ; i += 256) s[i] = media[(size_t)b * TQ + i] ? 1 : 0;
  __syncthreads();
  if (threadIdx.x == 0) {
    int acc = 0;
    for (int i = 0; i < TQ; ++i) { acc += s[i]; s[i] = acc; }
  }
  __syncthreads();
  for (int i = threadIdx.x; i < TQ; i += 256) qt[(size_t)b * TQ + i] = s[i];
}

__global__ __launch_bounds__(256) void k_transpose_bf16(const float* __restrict__ src,
                                                        __bf16* __restrict__ dst,
                                                        int rows, int cols) {
  int total = rows * cols;
  for (int idx = blockIdx.x * 256 + threadIdx.x; idx < total; idx += gridDim.x * 256) {
    int r = idx / cols, c = idx - r * cols;
    dst[(size_t)c * rows + r] = f2bf(src[idx]);
  }
}

__global__ __launch_bounds__(256) void k_cvt_bf16(const float* __restrict__ src,
                                                  __bf16* __restrict__ dst, int n) {
  for (int i = blockIdx.x * 256 + threadIdx.x; i < n; i += gridDim.x * 256)
    dst[i] = f2bf(src[i]);
}

__global__ __launch_bounds__(256) void k_layernorm(const float* __restrict__ x,
                                                   const float* __restrict__ g,
                                                   const float* __restrict__ be,
                                                   __bf16* __restrict__ out) {
  int row = blockIdx.x;
  const float* xr = x + (size_t)row * DIM;
  __shared__ float ssum[256], ssq[256];
  float s = 0.f, sq = 0.f;
  for (int i = threadIdx.x; i < DIM; i += 256) { float v = xr[i]; s += v; sq += v * v; }
  ssum[threadIdx.x] = s; ssq[threadIdx.x] = sq;
  __syncthreads();
  for (int off = 128; off > 0; off >>= 1) {
    if (threadIdx.x < off) {
      ssum[threadIdx.x] += ssum[threadIdx.x + off];
      ssq[threadIdx.x]  += ssq[threadIdx.x + off];
    }
    __syncthreads();
  }
  float mu   = ssum[0] * (1.0f / DIM);
  float var  = ssq[0] * (1.0f / DIM) - mu * mu;
  float rstd = rsqrtf(var + 1e-5f);
  __bf16* orow = out + (size_t)row * DIM;
  for (int i = threadIdx.x; i < DIM; i += 256)
    orow[i] = f2bf((xr[i] - mu) * rstd * g[i] + be[i]);
}

// ---------------------------------------------------------------------------
// GEMMs  (launch_bounds(256,1): minimum occupancy floor -> full register
// headroom so BOTH double-buffer bodies can hoist their 8-fragment clauses)
// ---------------------------------------------------------------------------

// Q projection: LNq[8192][1024] @ Wq -> Qs[b][h][i][d] bf16 (pre-scaled 1/sqrt(D))
__global__ __launch_bounds__(256, 1) void k_gemm_q(const __bf16* __restrict__ A,
                                                   const __bf16* __restrict__ Wt, // [HD][DIM]
                                                   __bf16* __restrict__ Qs) {
  int lane = threadIdx.x & 31, w = threadIdx.x >> 5;
  int n0 = (blockIdx.x & 7) << 6;                // 8 N-panels
  int m0 = (blockIdx.x >> 3) * 128 + w * 16;     // 64 row-blocks
  v8f acc[4];
#pragma unroll
  for (int f = 0; f < 4; ++f) acc[f] = zero8();
  gemm_panel(A, DIM, m0, Wt, DIM, HD, n0, w, lane, acc);
  int nlane = lane & 15, hh = lane >> 4;
#pragma unroll
  for (int f = 0; f < 4; ++f) {
    int n = n0 + f * 16 + nlane;
    int h = n >> 6, d = n & 63;
#pragma unroll
    for (int v = 0; v < 8; ++v) {
      int rowg = m0 + v + (hh << 3);
      int b = rowg >> 11, i = rowg & (TQ - 1);
      Qs[(((size_t)(b * H + h)) * TQ + i) * D + d] = f2bf(acc[f][v] * 0.125f);
    }
  }
}

// KV projection: kv_bf16[4096][1024] @ Wkv -> K [b][h][j][d], V^T [b][h][d][j]
__global__ __launch_bounds__(256, 1) void k_gemm_kv(const __bf16* __restrict__ A,
                                                    const __bf16* __restrict__ Wt, // [1024][1024]
                                                    __bf16* __restrict__ Ksd,
                                                    __bf16* __restrict__ Vtd) {
  int lane = threadIdx.x & 31, w = threadIdx.x >> 5;
  int n0 = (blockIdx.x & 15) << 6;               // 16 N-panels
  int m0 = (blockIdx.x >> 4) * 128 + w * 16;     // 32 row-blocks
  v8f acc[4];
#pragma unroll
  for (int f = 0; f < 4; ++f) acc[f] = zero8();
  gemm_panel(A, DLAT, m0, Wt, DLAT, DIM, n0, w, lane, acc);
  int nlane = lane & 15, hh = lane >> 4;
#pragma unroll
  for (int f = 0; f < 4; ++f) {
    int n = n0 + f * 16 + nlane;
#pragma unroll
    for (int v = 0; v < 8; ++v) {
      int rowg = m0 + v + (hh << 3);
      int b = rowg >> 10, j = rowg & (Jtot - 1);
      __bf16 val = f2bf(acc[f][v]);
      if (n < HD) {                                   // K half -> natural [j][d]
        int h = n >> 6, d = n & 63;
        Ksd[(((size_t)(b * H + h)) * Jtot + j) * D + d] = val;
      } else {                                        // V half -> transposed [d][j]
        int nn = n - HD, h = nn >> 6, d = nn & 63;
        Vtd[(((size_t)(b * H + h)) * D + d) * Jtot + j] = val;
      }
    }
  }
}

// Output projection: AO[8192][512] @ Wout -> out[8192][1024] f32
__global__ __launch_bounds__(256, 1) void k_gemm_out(const __bf16* __restrict__ A,
                                                     const __bf16* __restrict__ Wt, // [DIM][HD]
                                                     float* __restrict__ out) {
  int lane = threadIdx.x & 31, w = threadIdx.x >> 5;
  int n0 = (blockIdx.x & 15) << 6;               // 16 N-panels
  int m0 = (blockIdx.x >> 4) * 128 + w * 16;     // 64 row-blocks
  v8f acc[4];
#pragma unroll
  for (int f = 0; f < 4; ++f) acc[f] = zero8();
  gemm_panel(A, HD, m0, Wt, HD, DIM, n0, w, lane, acc);
  int nlane = lane & 15, hh = lane >> 4;
#pragma unroll
  for (int f = 0; f < 4; ++f)
#pragma unroll
    for (int v = 0; v < 8; ++v)
      out[(size_t)(m0 + v + (hh << 3)) * DIM + n0 + f * 16 + nlane] = acc[f][v];
}

// ---------------------------------------------------------------------------
// Fused masked attention (one wave per (b, h, 16-query tile)).
// Mask q_time == kv_time -> each row needs exactly one 64-latent chunk;
// only chunks in [tmin, tmax] of the tile are processed (usually one).
// ---------------------------------------------------------------------------
__global__ __launch_bounds__(256, 1) void k_attn(const __bf16* __restrict__ Qs,
                                                 const __bf16* __restrict__ Ks,
                                                 const __bf16* __restrict__ Vt,
                                                 const int* __restrict__ qt,
                                                 __bf16* __restrict__ AO) {
  __shared__ __bf16 Pbuf[8][16 * 64];
  int lane = threadIdx.x & 31;
  int w = threadIdx.x >> 5;
  int gw = blockIdx.x * 8 + w;
  int tile = gw & (TQ / 16 - 1);
  int bh = gw >> 7;
  int h = bh & (H - 1), b = bh >> 3;
  int i0 = tile << 4;

  const __bf16* Qb = Qs + (size_t)(b * H + h) * TQ * D;     // [i][d]
  const __bf16* Kb = Ks + (size_t)(b * H + h) * Jtot * D;   // [j][d]
  const __bf16* Vb = Vt + (size_t)(b * H + h) * D * Jtot;   // [d][j]

  v16bf aq0 = a_frag(Qb, D, i0, 0, lane);
  v16bf aq1 = a_frag(Qb, D, i0, 32, lane);

  int tv[16];
#pragma unroll
  for (int r = 0; r < 16; ++r) tv[r] = qt[b * TQ + i0 + r];   // wave-uniform
  int tmin = tv[0], tmax = tv[0];
#pragma unroll
  for (int r = 1; r < 16; ++r) { tmin = min(tmin, tv[r]); tmax = max(tmax, tv[r]); }

  v8f oacc[4];
#pragma unroll
  for (int f = 0; f < 4; ++f) oacc[f] = zero8();
  float rs[8];
#pragma unroll
  for (int v = 0; v < 8; ++v) rs[v] = 0.f;

  int nlane = lane & 15, hh = lane >> 4;
  __bf16* Pb = &Pbuf[w][0];

  int c0 = (tmin > 0 ? tmin : 1) - 1;
  int c1 = tmax - 1;                      // wave-uniform loop -> EXEC all ones at WMMA
  for (int c = c0; c <= c1; ++c) {
    int j0 = c * NLAT;
    // hoist ALL K and V fragments: one load clause, K first so the QK WMMAs
    // can proceed with the V loads still in flight
    v16bf bk[8], bv[8];
#pragma unroll
    for (int f = 0; f < 4; ++f) {
      bk[2 * f]     = b_frag(Kb, D, j0 + f * 16, 0, lane);
      bk[2 * f + 1] = b_frag(Kb, D, j0 + f * 16, 32, lane);
    }
#pragma unroll
    for (int f = 0; f < 4; ++f) {
      bv[2 * f]     = b_frag(Vb, Jtot, f * 16, j0, lane);
      bv[2 * f + 1] = b_frag(Vb, Jtot, f * 16, j0 + 32, lane);
    }
    v8f sim[4];
#pragma unroll
    for (int f = 0; f < 4; ++f) {
      v8f s = zero8();
      s = wmma_bf16(aq0, bk[2 * f], s);
      s = wmma_bf16(aq1, bk[2 * f + 1], s);
      sim[f] = s;
    }
    // per-row masked softmax over this chunk's 64 entries (half-wave shuffles)
#pragma unroll
    for (int v = 0; v < 8; ++v) {
      int m = v + (hh << 3);
      bool valid = (tv[m] == c + 1);
      float mx = fmaxf(fmaxf(sim[0][v], sim[1][v]), fmaxf(sim[2][v], sim[3][v]));
#pragma unroll
      for (int dd = 8; dd > 0; dd >>= 1) mx = fmaxf(mx, __shfl_xor(mx, dd, 32));
      float p0 = valid ? __expf(sim[0][v] - mx) : 0.f;
      float p1 = valid ? __expf(sim[1][v] - mx) : 0.f;
      float p2 = valid ? __expf(sim[2][v] - mx) : 0.f;
      float p3 = valid ? __expf(sim[3][v] - mx) : 0.f;
      float sum = p0 + p1 + p2 + p3;
#pragma unroll
      for (int dd = 8; dd > 0; dd >>= 1) sum += __shfl_xor(sum, dd, 32);
      rs[v] += sum;                       // exactly one chunk contributes per row
      Pb[m * 64 +  0 + nlane] = f2bf(p0);
      Pb[m * 64 + 16 + nlane] = f2bf(p1);
      Pb[m * 64 + 32 + nlane] = f2bf(p2);
      Pb[m * 64 + 48 + nlane] = f2bf(p3);
    }
    asm volatile("" ::: "memory");        // order wave-local LDS stores vs A-frag reads
    v16bf ap0 = a_frag(Pb, 64, 0, 0, lane);
    v16bf ap1 = a_frag(Pb, 64, 0, 32, lane);
#pragma unroll
    for (int f = 0; f < 4; ++f) {
      oacc[f] = wmma_bf16(ap0, bv[2 * f], oacc[f]);
      oacc[f] = wmma_bf16(ap1, bv[2 * f + 1], oacc[f]);
    }
    asm volatile("" ::: "memory");
  }

#pragma unroll
  for (int v = 0; v < 8; ++v) {
    float inv = rs[v] > 0.f ? 1.f / rs[v] : 0.f;   // q_time==0 rows -> zero
    int i = i0 + v + (hh << 3);
#pragma unroll
    for (int f = 0; f < 4; ++f)
      AO[((size_t)(b * TQ + i)) * HD + h * D + f * 16 + nlane] = f2bf(oacc[f][v] * inv);
  }
}

// ---------------------------------------------------------------------------
// Host launcher
// ---------------------------------------------------------------------------
extern "C" void kernel_launch(void* const* d_in, const int* in_sizes, int n_in,
                              void* d_out, int out_size, void* d_ws, size_t ws_size,
                              hipStream_t stream) {
  (void)in_sizes; (void)n_in; (void)out_size; (void)ws_size;
  const float*   qo    = (const float*)d_in[0];
  const float*   kvo   = (const float*)d_in[1];
  const uint8_t* media = (const uint8_t*)d_in[2];   // numpy bool -> 1 byte
  const float*   ln_g  = (const float*)d_in[3];
  const float*   ln_b  = (const float*)d_in[4];
  const float*   Wq    = (const float*)d_in[5];
  const float*   Wkv   = (const float*)d_in[6];
  const float*   Wout  = (const float*)d_in[7];
  float* out = (float*)d_out;

  char* ws = (char*)d_ws;
  size_t off = 0;
  auto alloc = [&](size_t bytes) -> char* {
    char* p = ws + off;
    off = (off + bytes + 255) & ~(size_t)255;
    return p;
  };
  int*    qt     = (int*)   alloc((size_t)Bc * TQ * 4);
  __bf16* Wq_t   = (__bf16*)alloc((size_t)HD * DIM * 2);        // [512][1024]
  __bf16* Wkv_t  = (__bf16*)alloc((size_t)DIM * DLAT * 2);      // [1024][1024]
  __bf16* Wout_t = (__bf16*)alloc((size_t)DIM * HD * 2);        // [1024][512]
  __bf16* LNq    = (__bf16*)alloc((size_t)Bc * TQ * DIM * 2);   // [8192][1024]
  __bf16* kvbf   = (__bf16*)alloc((size_t)Bc * Jtot * DLAT * 2);// [4096][1024]
  __bf16* Qs     = (__bf16*)alloc((size_t)Bc * H * TQ * D * 2);
  __bf16* Ksd    = (__bf16*)alloc((size_t)Bc * H * Jtot * D * 2);
  __bf16* Vtd    = (__bf16*)alloc((size_t)Bc * H * D * Jtot * 2);
  __bf16* AO     = (__bf16*)alloc((size_t)Bc * TQ * HD * 2);    // [8192][512]

  k_qtime<<<Bc, 256, 0, stream>>>(media, qt);
  k_transpose_bf16<<<512,  256, 0, stream>>>(Wq,   Wq_t,   DIM, HD);
  k_transpose_bf16<<<1024, 256, 0, stream>>>(Wkv,  Wkv_t,  DLAT, DIM);
  k_transpose_bf16<<<512,  256, 0, stream>>>(Wout, Wout_t, HD, DIM);
  k_cvt_bf16<<<2048, 256, 0, stream>>>(kvo, kvbf, Bc * Jtot * DLAT);
  k_layernorm<<<Bc * TQ, 256, 0, stream>>>(qo, ln_g, ln_b, LNq);
  k_gemm_q  <<<512,  256, 0, stream>>>(LNq,  Wq_t,  Qs);
  k_gemm_kv <<<512,  256, 0, stream>>>(kvbf, Wkv_t, Ksd, Vtd);
  k_attn    <<<512,  256, 0, stream>>>(Qs, Ksd, Vtd, qt, AO);
  k_gemm_out<<<1024, 256, 0, stream>>>(AO, Wout_t, out);
}
